// BDF_65979287601426
// MI455X (gfx1250) — matrix operand order repaired and checked by reading, not compiled, not run on hardware
//
#include <hip/hip_runtime.h>

#define HDIM 128
#define HWSZ (HDIM*HDIM)      // 16384
#define BATCH 8
#define CCH 64
#define TSZ (BATCH*CCH*HWSZ)  // 8388608 floats per tensor

typedef __attribute__((ext_vector_type(16))) __bf16 v16bf;
typedef __attribute__((ext_vector_type(8)))  float  v8f;
typedef __attribute__((ext_vector_type(8)))  int    v8i;

__device__ __forceinline__ unsigned pack_bf16(float a, float b) {
  unsigned ua = __float_as_uint(a), ub = __float_as_uint(b);
  ua = (ua + 0x7FFFu + ((ua >> 16) & 1u)) >> 16;
  ub = (ub + 0x7FFFu + ((ub >> 16) & 1u)) >> 16;
  return (ub << 16) | (ua & 0xFFFFu);
}
__device__ __forceinline__ float sigmoidf_(float x) { return 1.f / (1.f + __expf(-x)); }

// ---------------------------------------------------------------------------
// Depthwise conv (kh x kw, dilation d) + BN + ReLU
// ---------------------------------------------------------------------------
__global__ __launch_bounds__(256) void k_dwconv(
    const float* __restrict__ X, float* __restrict__ Y, const float* __restrict__ Wt,
    const float* __restrict__ bg, const float* __restrict__ bb,
    const float* __restrict__ bm, const float* __restrict__ bv,
    float eps, int kh, int kw, int dil) {
  long i = (long)blockIdx.x * 256 + threadIdx.x;
  if (i >= (long)TSZ) return;
  int p = (int)(i % HWSZ); int c = (int)((i / HWSZ) % CCH); int b = (int)(i / ((long)HWSZ * CCH));
  int y = p / HDIM, x = p % HDIM;
  int ph = dil * (kh - 1) / 2, pw = dil * (kw - 1) / 2;
  const float* xb = X + ((long)b * CCH + c) * HWSZ;
  float acc = 0.f;
  for (int ki = 0; ki < kh; ki++) {
    int yy = y + ki * dil - ph;
    if (yy < 0 || yy >= HDIM) continue;
    for (int kj = 0; kj < kw; kj++) {
      int xx = x + kj * dil - pw;
      if (xx < 0 || xx >= HDIM) continue;
      acc += Wt[(c * kh + ki) * kw + kj] * xb[yy * HDIM + xx];
    }
  }
  float sc = bg[c] * rsqrtf(bv[c] + eps);
  float r = (acc - bm[c]) * sc + bb[c];
  Y[i] = fmaxf(r, 0.f);
}

// ---------------------------------------------------------------------------
// Global pooling stats: pooled[b, q*64+c] = w0*mean + w1*max + w2*std(ddof=1)
// ---------------------------------------------------------------------------
__global__ __launch_bounds__(256) void k_pool_stats(
    const float* __restrict__ x, const float* __restrict__ o1,
    const float* __restrict__ o2, const float* __restrict__ o3,
    const float* __restrict__ pool_w, float* __restrict__ pooled) {
  __shared__ float ss[256], sq[256], sm_[256];
  int t = threadIdx.x, blk = blockIdx.x;
  int b = blk >> 8, ch4 = blk & 255, q = ch4 >> 6, c = ch4 & 63;
  const float* src = (q == 0) ? x : (q == 1) ? o1 : (q == 2) ? o2 : o3;
  src += ((long)b * CCH + c) * HWSZ;
  float s = 0.f, s2 = 0.f, mx = -3.4e38f;
  for (int i = t; i < HWSZ; i += 256) { float v = src[i]; s += v; s2 += v * v; mx = fmaxf(mx, v); }
  ss[t] = s; sq[t] = s2; sm_[t] = mx; __syncthreads();
  for (int o = 128; o > 0; o >>= 1) {
    if (t < o) { ss[t] += ss[t + o]; sq[t] += sq[t + o]; sm_[t] = fmaxf(sm_[t], sm_[t + o]); }
    __syncthreads();
  }
  if (t == 0) {
    float n = (float)HWSZ;
    float mean = ss[0] / n;
    float var = fmaxf(0.f, (sq[0] - ss[0] * ss[0] / n) / (n - 1.f));
    float w0 = sigmoidf_(pool_w[0]), w1 = sigmoidf_(pool_w[1]), w2 = sigmoidf_(pool_w[2]);
    pooled[blk] = w0 * mean + w1 * sm_[0] + w2 * sqrtf(var);
  }
}

// ---------------------------------------------------------------------------
// Channel attention: group-compress + double ECA conv1d + fc + gate pw-block
// ---------------------------------------------------------------------------
__global__ __launch_bounds__(64) void k_channel_att(
    const float* __restrict__ pooled, const float* __restrict__ gc_w, const float* __restrict__ gc_b,
    const float* __restrict__ e1, const float* __restrict__ e2,
    const float* __restrict__ fc_w, const float* __restrict__ fc_b,
    const float* __restrict__ gW, const float* __restrict__ gB,
    const float* __restrict__ gg, const float* __restrict__ gb2,
    const float* __restrict__ gm, const float* __restrict__ gv,
    float* __restrict__ gate) {
  __shared__ float sh[256], t1[128], t2[64], sv[64];
  int c = threadIdx.x, b = blockIdx.x;
  float cag = gc_b[c];
  for (int f = 0; f < 4; f++) {
    float pv = pooled[b * 256 + f * 64 + c];
    sh[c * 4 + f] = pv;
    cag += pv * gc_w[c * 4 + f];
  }
  __syncthreads();
  for (int i = c; i < 128; i += 64) {
    float a = 0.f;
    for (int k = 0; k < 5; k++) { int j = 2 * i + k - 2; if (j >= 0 && j < 256) a += e1[k] * sh[j]; }
    t1[i] = a;
  }
  __syncthreads();
  {
    float a = 0.f;
    for (int k = 0; k < 5; k++) { int j = 2 * c + k - 2; if (j >= 0 && j < 128) a += e2[k] * t1[j]; }
    t2[c] = a;
  }
  __syncthreads();
  {
    float a = fc_b[c];
    for (int j = 0; j < 64; j++) a += t2[j] * fc_w[c * 64 + j];
    sv[c] = cag + a;
  }
  __syncthreads();
  {
    float a = gB[c];
    for (int i = 0; i < 64; i++) a += gW[c * 64 + i] * sv[i];
    float sc = gg[c] * rsqrtf(gv[c] + 1e-5f);
    gate[b * 64 + c] = sigmoidf_((a - gm[c]) * sc + gb2[c]);
  }
}

// ---------------------------------------------------------------------------
// Spatial group-compress: spa = sum_f gcs_w[c,f] * combined[f] + gcs_b
// ---------------------------------------------------------------------------
__global__ __launch_bounds__(256) void k_spa(
    const float* __restrict__ x, const float* __restrict__ o1,
    const float* __restrict__ o2, const float* __restrict__ o3,
    const float* __restrict__ gcs_w, const float* __restrict__ gcs_b, float* __restrict__ spa) {
  long i = (long)blockIdx.x * 256 + threadIdx.x;
  if (i >= (long)TSZ) return;
  int c = (int)((i / HWSZ) % CCH);
  spa[i] = gcs_w[c * 4 + 0] * x[i] + gcs_w[c * 4 + 1] * o1[i] +
           gcs_w[c * 4 + 2] * o2[i] + gcs_w[c * 4 + 3] * o3[i] + gcs_b[c];
}

// mean/max over channels -> (b,2,h,w)
__global__ __launch_bounds__(256) void k_sa_reduce(const float* __restrict__ spa, float* __restrict__ sain) {
  long i = (long)blockIdx.x * 256 + threadIdx.x;
  if (i >= (long)BATCH * HWSZ) return;
  int b = (int)(i / HWSZ), p = (int)(i % HWSZ);
  const float* sp_ = spa + (long)b * CCH * HWSZ + p;
  float s = 0.f, m = -3.4e38f;
  for (int c = 0; c < 64; c++) { float v = sp_[(long)c * HWSZ]; s += v; m = fmaxf(m, v); }
  sain[(long)(b * 2 + 0) * HWSZ + p] = s / 64.f;
  sain[(long)(b * 2 + 1) * HWSZ + p] = m;
}

// 7x7 conv (2 -> 1 channels, pad 3) + sigmoid
__global__ __launch_bounds__(256) void k_sa_conv7(
    const float* __restrict__ sain, const float* __restrict__ w, float* __restrict__ mask) {
  long i = (long)blockIdx.x * 256 + threadIdx.x;
  if (i >= (long)BATCH * HWSZ) return;
  int b = (int)(i / HWSZ), p = (int)(i % HWSZ);
  int y = p / HDIM, x = p % HDIM;
  float acc = 0.f;
  for (int ch = 0; ch < 2; ch++) {
    const float* sb = sain + (long)(b * 2 + ch) * HWSZ;
    for (int ki = 0; ki < 7; ki++) {
      int yy = y + ki - 3; if (yy < 0 || yy >= HDIM) continue;
      for (int kj = 0; kj < 7; kj++) {
        int xx = x + kj - 3; if (xx < 0 || xx >= HDIM) continue;
        acc += w[ch * 49 + ki * 7 + kj] * sb[yy * HDIM + xx];
      }
    }
  }
  mask[i] = sigmoidf_(acc);
}

// ---------------------------------------------------------------------------
// Generic WMMA pointwise conv: Y = act(bn(W @ (X*scaleC*scaleP) + bias)) [+res]
// Block = 128 threads (4 waves), each wave computes 64(co) x 16(pix).
// ---------------------------------------------------------------------------
__global__ __launch_bounds__(128) void k_pw_wmma(
    const float* __restrict__ X, float* __restrict__ Y, const float* __restrict__ W,
    const float* __restrict__ bias,
    const float* __restrict__ bng, const float* __restrict__ bnb,
    const float* __restrict__ bnm, const float* __restrict__ bnv,
    float eps, int actmode,                  // 0=linear, 1=bn+relu, 2=bn+sigmoid
    const float* __restrict__ scaleC,        // [b][64]  or null
    const float* __restrict__ scaleP,        // [b][HW]  or null
    const float* __restrict__ resid,         // [b][64][HW] or null
    int HW) {
  __shared__ unsigned Wp[32][64];            // bf16 pair-packed weights (A frags)
  __shared__ unsigned Xp[4][32][16];         // per-wave bf16 pair-packed inputs (B frags)
  int t = threadIdx.x, wave = t >> 5, lane = t & 31;
  // Stage W: Wp[sp][co] = pack(W[co][2sp], W[co][2sp+1])
  for (int i = t; i < 32 * 64; i += 128) {
    int sp = i >> 6, co = i & 63;
    Wp[sp][co] = pack_bf16(W[co * 64 + 2 * sp], W[co * 64 + 2 * sp + 1]);
  }
  long pixw = (long)blockIdx.x * 64 + wave * 16;
  int b = (int)(pixw / HW);
  int p0 = (int)(pixw % HW);
  const float* xb = X + ((long)b * CCH) * HW;
  for (int i = lane; i < 32 * 16; i += 32) {
    int cp = i >> 4, pp = i & 15, p = p0 + pp;
    float x0 = xb[(long)(2 * cp) * HW + p];
    float x1 = xb[(long)(2 * cp + 1) * HW + p];
    if (scaleC) { x0 *= scaleC[b * 64 + 2 * cp]; x1 *= scaleC[b * 64 + 2 * cp + 1]; }
    if (scaleP) { float s = scaleP[(long)b * HW + p]; x0 *= s; x1 *= s; }
    Xp[wave][cp][pp] = pack_bf16(x0, x1);
  }
  __syncthreads();
  int g = lane >> 4, ln = lane & 15;
  v8f acc[4] = {};
  for (int kk = 0; kk < 2; kk++) {
    union { int i[8]; v8i v; } fb;
    for (int v = 0; v < 8; v++) fb.i[v] = (int)Xp[wave][kk * 16 + g * 8 + v][ln];
    v16bf B = __builtin_bit_cast(v16bf, fb.v);
    for (int tm = 0; tm < 4; tm++) {
      union { int i[8]; v8i v; } fa;
      int co = tm * 16 + ln;
      for (int v = 0; v < 8; v++) {
        int spl = g * 8 + v;
        fa.i[v] = (int)Wp[kk * 16 + spl][co];
      }
      v16bf A = __builtin_bit_cast(v16bf, fa.v);
      acc[tm] = __builtin_amdgcn_wmma_f32_16x16x32_bf16(false, A, false, B, (short)0, acc[tm], false, false);
    }
  }
  int p = p0 + ln;
  for (int tm = 0; tm < 4; tm++) {
    for (int r = 0; r < 8; r++) {
      int co = tm * 16 + g * 8 + r;
      float y = acc[tm][r] + bias[co];
      if (actmode) {
        float sc = bng[co] * rsqrtf(bnv[co] + eps);
        y = (y - bnm[co]) * sc + bnb[co];
        y = (actmode == 1) ? fmaxf(y, 0.f) : sigmoidf_(y);
      }
      long oi = ((long)b * CCH + co) * HW + p;
      if (resid) y += resid[oi];
      Y[oi] = y;
    }
  }
}

// ---------------------------------------------------------------------------
// PSP adaptive max pool to grids 7/5/3/1 -> pool84 (b,64,84)
// ---------------------------------------------------------------------------
__global__ __launch_bounds__(256) void k_psp(const float* __restrict__ feat, float* __restrict__ pool84) {
  int i = blockIdx.x * 256 + threadIdx.x;
  if (i >= BATCH * CCH * 84) return;
  int s = i % 84, bc = i / 84, g, si;
  if (s < 49)      { g = 7; si = s; }
  else if (s < 74) { g = 5; si = s - 49; }
  else if (s < 83) { g = 3; si = s - 74; }
  else             { g = 1; si = 0; }
  int gi = si / g, gj = si % g;
  int r0 = gi * HDIM / g, r1 = ((gi + 1) * HDIM + g - 1) / g;
  int c0 = gj * HDIM / g, c1 = ((gj + 1) * HDIM + g - 1) / g;
  const float* f = feat + (long)bc * HWSZ;
  float m = -3.4e38f;
  for (int r = r0; r < r1; r++)
    for (int cc = c0; cc < c1; cc++) m = fmaxf(m, f[r * HDIM + cc]);
  pool84[i] = m;
}

// K projection -> bf16 pair-packed over channel dim: Kp[b][cp][s96]
__global__ __launch_bounds__(256) void k_proj_K(
    const float* __restrict__ pool84, const float* __restrict__ wk,
    const float* __restrict__ bk, unsigned* __restrict__ Kp) {
  int i = blockIdx.x * 256 + threadIdx.x;
  if (i >= BATCH * 32 * 96) return;
  int s = i % 96, bcp = i / 96, cp = bcp % 32, b = bcp / 32;
  float k0 = 0.f, k1 = 0.f;
  if (s < 84) {
    k0 = bk[2 * cp]; k1 = bk[2 * cp + 1];
    const float* pb = pool84 + (long)b * CCH * 84;
    for (int ci = 0; ci < 64; ci++) {
      float pv = pb[ci * 84 + s];
      k0 += wk[(2 * cp) * 64 + ci] * pv;
      k1 += wk[(2 * cp + 1) * 64 + ci] * pv;
    }
  }
  Kp[i] = pack_bf16(k0, k1);
}

// V projection -> bf16 pair-packed over s dim: Vp[b][c][sp48]
__global__ __launch_bounds__(256) void k_proj_V(
    const float* __restrict__ pool84, const float* __restrict__ wv,
    const float* __restrict__ bv, unsigned* __restrict__ Vp) {
  int i = blockIdx.x * 256 + threadIdx.x;
  if (i >= BATCH * CCH * 48) return;
  int sp = i % 48, bc = i / 48, c = bc % 64, b = bc / 64;
  const float* pb = pool84 + (long)b * CCH * 84;
  float v0 = 0.f, v1 = 0.f;
  int s0 = 2 * sp, s1 = 2 * sp + 1;
  if (s0 < 84) { v0 = bv[c]; for (int ci = 0; ci < 64; ci++) v0 += wv[c * 64 + ci] * pb[ci * 84 + s0]; }
  if (s1 < 84) { v1 = bv[c]; for (int ci = 0; ci < 64; ci++) v1 += wv[c * 64 + ci] * pb[ci * 84 + s1]; }
  Vp[i] = pack_bf16(v0, v1);
}

// ---------------------------------------------------------------------------
// Fused attention: one wave handles 16 query rows.
// logits(16x96)=Q^T K via WMMA, softmax(84) in LDS, ctx(64x16)=V sim^T via WMMA
// ---------------------------------------------------------------------------
__global__ __launch_bounds__(32) void k_attn(
    const float* __restrict__ Q, const unsigned* __restrict__ Kp,
    const unsigned* __restrict__ Vp, float* __restrict__ Ctx, int HW) {
  __shared__ float    qs[64][16];
  __shared__ float    ls[16][96];
  __shared__ unsigned simp[48][16];
  int lane = threadIdx.x;
  long blk = blockIdx.x;
  int b = (int)(blk / (HW / 16)), n0 = (int)(blk % (HW / 16)) * 16;
  const float* qb = Q + ((long)b * CCH) * HW + n0;
  for (int i = lane; i < 64 * 16; i += 32) {
    int ci = i >> 4, pp = i & 15;
    qs[ci][pp] = qb[(long)ci * HW + pp];
  }
  __syncthreads();
  int g = lane >> 4, ln = lane & 15;
  const unsigned* kb = Kp + (long)b * 32 * 96;
  for (int st = 0; st < 6; st++) {
    v8f acc = {};
    for (int kk = 0; kk < 2; kk++) {
      union { int i[8]; v8i v; } fa, fb;
      for (int v = 0; v < 8; v++) {
        int spl = g * 8 + v;
        int k0 = kk * 32 + 2 * spl;
        fa.i[v] = (int)pack_bf16(qs[k0][ln], qs[k0 + 1][ln]);
        fb.i[v] = (int)kb[(kk * 16 + g * 8 + v) * 96 + st * 16 + ln];
      }
      acc = __builtin_amdgcn_wmma_f32_16x16x32_bf16(
          false, __builtin_bit_cast(v16bf, fa.v), false, __builtin_bit_cast(v16bf, fb.v),
          (short)0, acc, false, false);
    }
    for (int r = 0; r < 8; r++) ls[g * 8 + r][st * 16 + ln] = acc[r];
  }
  __syncthreads();
  if (lane < 16) {
    float mx = -3.4e38f;
    for (int s = 0; s < 84; s++) mx = fmaxf(mx, ls[lane][s]);
    float den = 0.f;
    for (int s = 0; s < 84; s++) { float e = __expf(ls[lane][s] - mx); ls[lane][s] = e; den += e; }
    float inv = 1.f / den;
    for (int sp = 0; sp < 48; sp++) {
      int s0 = 2 * sp, s1 = 2 * sp + 1;
      float v0 = (s0 < 84) ? ls[lane][s0] * inv : 0.f;
      float v1 = (s1 < 84) ? ls[lane][s1] * inv : 0.f;
      simp[sp][lane] = pack_bf16(v0, v1);
    }
  }
  __syncthreads();
  const unsigned* vb = Vp + (long)b * CCH * 48;
  float* cb = Ctx + ((long)b * CCH) * HW + n0;
  for (int tm = 0; tm < 4; tm++) {
    v8f acc = {};
    for (int kk = 0; kk < 3; kk++) {
      union { int i[8]; v8i v; } fa, fb;
      int c = tm * 16 + ln;
      for (int v = 0; v < 8; v++) {
        int spl = g * 8 + v;
        fa.i[v] = (int)vb[c * 48 + kk * 16 + spl];
        fb.i[v] = (int)simp[kk * 16 + g * 8 + v][ln];
      }
      acc = __builtin_amdgcn_wmma_f32_16x16x32_bf16(
          false, __builtin_bit_cast(v16bf, fa.v), false, __builtin_bit_cast(v16bf, fb.v),
          (short)0, acc, false, false);
    }
    for (int r = 0; r < 8; r++) cb[(long)(tm * 16 + g * 8 + r) * HW + ln] = acc[r];
  }
}

// att = f + ctx_c*a + a + ctx_a*c + c
__global__ __launch_bounds__(256) void k_combine(
    const float* __restrict__ f, const float* __restrict__ ctxc, const float* __restrict__ a,
    const float* __restrict__ ctxa, const float* __restrict__ c, float* __restrict__ out) {
  long i = (long)blockIdx.x * 256 + threadIdx.x;
  if (i >= (long)TSZ) return;
  out[i] = f[i] + ctxc[i] * a[i] + a[i] + ctxa[i] * c[i] + c[i];
}

// ---------------------------------------------------------------------------
extern "C" void kernel_launch(void* const* d_in, const int* in_sizes, int n_in,
                              void* d_out, int out_size, void* d_ws, size_t ws_size,
                              hipStream_t stream) {
  (void)in_sizes; (void)n_in; (void)out_size; (void)ws_size;
#define F(i) ((const float*)d_in[i])
  const float* x = F(0);
  float* ws = (float*)d_ws;
  float* B[10];
  for (int i = 0; i < 10; i++) B[i] = ws + (size_t)i * TSZ;
  float* sm = ws + (size_t)10 * TSZ;
  float* pooled256 = sm;               sm += 2048;
  float* gate      = sm;               sm += 512;
  float* sain      = sm;               sm += (size_t)2 * BATCH * HWSZ;
  float* mask      = sm;               sm += (size_t)BATCH * HWSZ;
  float* pool84    = sm;               sm += (size_t)BATCH * CCH * 84;
  unsigned* Kp = (unsigned*)sm;        sm += (size_t)BATCH * 32 * 96;
  unsigned* Vp = (unsigned*)sm;

  dim3 blk(256);
  int gEl  = (TSZ + 255) / 256;
  int gBP  = (BATCH * HWSZ + 255) / 256;
  int gPW  = (BATCH * HWSZ) / 64;
  int gPSP = (BATCH * CCH * 84 + 255) / 256;
  const float* NUL = nullptr;

  // dilated depthwise convs (Conv wrapper BN eps=1e-3)
  k_dwconv<<<gEl, blk, 0, stream>>>(x, B[0], F(1),  F(2),  F(3),  F(4),  F(5),  1e-3f, 3, 3, 1);
  k_dwconv<<<gEl, blk, 0, stream>>>(x, B[1], F(6),  F(7),  F(8),  F(9),  F(10), 1e-3f, 3, 3, 2);
  k_dwconv<<<gEl, blk, 0, stream>>>(x, B[2], F(11), F(12), F(13), F(14), F(15), 1e-3f, 3, 3, 3);
  // pooling stats + channel attention gate
  k_pool_stats<<<BATCH * 256, blk, 0, stream>>>(x, B[0], B[1], B[2], F(16), pooled256);
  k_channel_att<<<BATCH, 64, 0, stream>>>(pooled256, F(19), F(20), F(17), F(18), F(21), F(22),
                                          F(23), F(24), F(25), F(26), F(27), F(28), gate);
  // spatial attention
  k_spa<<<gEl, blk, 0, stream>>>(x, B[0], B[1], B[2], F(29), F(30), B[3]);
  k_pw_wmma<<<gPW, 128, 0, stream>>>(B[3], B[4], F(31), F(32), F(33), F(34), F(35), F(36),
                                     1e-5f, 1, NUL, NUL, NUL, HWSZ);           // pw(spa)
  k_sa_reduce<<<gBP, blk, 0, stream>>>(B[4], sain);
  k_sa_conv7<<<gBP, blk, 0, stream>>>(sain, F(37), mask);
  k_pw_wmma<<<gPW, 128, 0, stream>>>(x, B[5], F(38), F(39), F(40), F(41), F(42), F(43),
                                     1e-5f, 1, NUL, mask, NUL, HWSZ);          // x_sa = pw2(x*mask)
  // a_feat chain (x_ca = x*gate fused into pw3)
  k_pw_wmma<<<gPW, 128, 0, stream>>>(x, B[3], F(54), F(55), F(56), F(57), F(58), F(59),
                                     1e-5f, 1, gate, NUL, NUL, HWSZ);          // pw3
  k_dwconv<<<gEl, blk, 0, stream>>>(B[3], B[4], F(44), F(45), F(46), F(47), F(48), 1e-5f, 1, 3, 1);
  k_dwconv<<<gEl, blk, 0, stream>>>(B[4], B[3], F(49), F(50), F(51), F(52), F(53), 1e-5f, 3, 1, 1); // a_feat=B3
  // c_feat chain
  k_pw_wmma<<<gPW, 128, 0, stream>>>(B[5], B[4], F(70), F(71), F(72), F(73), F(74), F(75),
                                     1e-5f, 1, NUL, NUL, NUL, HWSZ);           // pw4
  k_dwconv<<<gEl, blk, 0, stream>>>(B[4], B[6], F(60), F(61), F(62), F(63), F(64), 1e-5f, 1, 3, 1);
  k_dwconv<<<gEl, blk, 0, stream>>>(B[6], B[4], F(65), F(66), F(67), F(68), F(69), 1e-5f, 3, 1, 1); // c_feat=B4
  // f chain
  k_pw_wmma<<<gPW, 128, 0, stream>>>(x, B[5], F(86), F(87), F(88), F(89), F(90), F(91),
                                     1e-5f, 1, NUL, NUL, NUL, HWSZ);           // pw5
  k_dwconv<<<gEl, blk, 0, stream>>>(B[5], B[6], F(76), F(77), F(78), F(79), F(80), 1e-5f, 1, 3, 1);
  k_dwconv<<<gEl, blk, 0, stream>>>(B[6], B[5], F(81), F(82), F(83), F(84), F(85), 1e-5f, 3, 1, 1);
  k_pw_wmma<<<gPW, 128, 0, stream>>>(B[5], B[6], F(92), F(93), F(94), F(95), F(96), F(97),
                                     1e-5f, 1, NUL, NUL, NUL, HWSZ);           // f=B6
  // FDB branch A (kv = a_feat)
  k_pw_wmma<<<gPW, 128, 0, stream>>>(B[6], B[7], F(98), F(106), NUL, NUL, NUL, NUL,
                                     0.f, 0, NUL, NUL, NUL, HWSZ);             // q_a
  k_psp<<<gPSP, blk, 0, stream>>>(B[3], pool84);
  k_proj_K<<<(BATCH * 32 * 96 + 255) / 256, blk, 0, stream>>>(pool84, F(102), F(108), Kp);
  k_proj_V<<<(BATCH * CCH * 48 + 255) / 256, blk, 0, stream>>>(pool84, F(103), F(109), Vp);
  k_attn<<<BATCH * (HWSZ / 16), 32, 0, stream>>>(B[7], Kp, Vp, B[8], HWSZ);
  k_pw_wmma<<<gPW, 128, 0, stream>>>(B[8], B[9], F(100), F(112), NUL, NUL, NUL, NUL,
                                     0.f, 0, NUL, NUL, NUL, HWSZ);             // ctx_a=B9
  // FDB branch C (kv = c_feat)
  k_pw_wmma<<<gPW, 128, 0, stream>>>(B[6], B[7], F(99), F(107), NUL, NUL, NUL, NUL,
                                     0.f, 0, NUL, NUL, NUL, HWSZ);             // q_c
  k_psp<<<gPSP, blk, 0, stream>>>(B[4], pool84);
  k_proj_K<<<(BATCH * 32 * 96 + 255) / 256, blk, 0, stream>>>(pool84, F(104), F(110), Kp);
  k_proj_V<<<(BATCH * CCH * 48 + 255) / 256, blk, 0, stream>>>(pool84, F(105), F(111), Vp);
  k_attn<<<BATCH * (HWSZ / 16), 32, 0, stream>>>(B[7], Kp, Vp, B[8], HWSZ);
  k_pw_wmma<<<gPW, 128, 0, stream>>>(B[8], B[8], F(101), F(113), NUL, NUL, NUL, NUL,
                                     0.f, 0, NUL, NUL, NUL, HWSZ);             // ctx_c=B8 (in-place safe)
  // att = f + ctx_c*a + a + ctx_a*c + c ; then out pw-block + residual x
  k_combine<<<gEl, blk, 0, stream>>>(B[6], B[8], B[3], B[9], B[4], B[0]);
  k_pw_wmma<<<gPW, 128, 0, stream>>>(B[0], (float*)d_out, F(114), F(115), F(116), F(117), F(118), F(119),
                                     1e-5f, 1, NUL, NUL, x, HWSZ);
#undef F
}
